// TemporalSAE_42623255446003
// MI455X (gfx1250) — matrix operand order repaired and checked by reading, not compiled, not run on hardware
//
#include <hip/hip_runtime.h>

#define DEV_INLINE __device__ __forceinline__

typedef __bf16 bf16;
typedef __bf16 v16bf  __attribute__((ext_vector_type(16)));
typedef __bf16 bf16x8 __attribute__((ext_vector_type(8)));
typedef __bf16 bf16x4 __attribute__((ext_vector_type(4)));
typedef float  v8f    __attribute__((ext_vector_type(8)));
typedef float  f32x4  __attribute__((ext_vector_type(4)));

// ---------------- problem constants ----------------
constexpr int kB      = 8;
constexpr int kT      = 512;
constexpr int kDIMIN  = 768;
constexpr int kWIDTH  = 8192;
constexpr int kNHEADS = 8;
constexpr int kNEMBDS = 128;   // qk bottleneck
constexpr int kDHQK   = 16;
constexpr int kDHV    = 1024;
constexpr int kNTOK   = kB * kT;                 // 4096
constexpr float kLAM  = 1.0f / (4.0f * kDIMIN);  // 1/3072
constexpr float kEPS  = 1e-6f;

// ---------------- GEMM tiling ----------------
// Block tile 128x128, 8 wave32s of 32(M)x64(N) -> 8 WMMA per wave per K-tile.
constexpr int BM = 128, BN = 128, TK = 32;
constexpr int LDSK = TK + 8;   // 40 bf16 -> 80B row stride (16B aligned)

enum { A_PLAIN = 0, A_SUBB = 1, A_SHIFT = 2, A_COMBINE = 3 };

// ---------------- WMMA helpers (CDNA5 wave32, 16x16x32 bf16) ----------------
DEV_INLINE v8f wmma_bf16(v16bf a, v16bf b, v8f c) {
  // (neg_a, A, neg_b, B, c_mod, C, reuse_a, reuse_b)
  return __builtin_amdgcn_wmma_f32_16x16x32_bf16(false, a, false, b, (short)0, c,
                                                 false, false);
}

// A matrix 16x32 bf16 fragment (ISA 7.12.2): lanes0-15 kbase=0, lanes16-31
// kbase=8; element pairs cover K=[kb,kb+8) then K=[16+kb,16+kb+8).
// Tile stored row-major [m][k] (k-contiguous) -> two 16B LDS loads.
DEV_INLINE v16bf load_a_frag(const bf16* As, int m0, int lane) {
  const int m  = m0 + (lane & 15);
  const int kb = (lane >> 4) * 8;
  const bf16x8 lo = *(const bf16x8*)(&As[m * LDSK + kb]);
  const bf16x8 hi = *(const bf16x8*)(&As[m * LDSK + kb + 16]);
  return __builtin_shufflevector(lo, hi, 0, 1, 2, 3, 4, 5, 6, 7, 8, 9, 10, 11,
                                 12, 13, 14, 15);
}

// B matrix 32x16 bf16 fragment: same K order as A (kb=0/8, K=[kb,kb+8) then [16+kb,16+kb+8)),
// N = lane%16. Tile stored n-major [n][k] (k-contiguous) -> two 16B LDS loads.
DEV_INLINE v16bf load_b_frag(const bf16* Bs, int n0, int lane) {
  const int n  = n0 + (lane & 15);
  const int kb = (lane >> 4) * 8;
  const bf16x8 lo = *(const bf16x8*)(&Bs[n * LDSK + kb]);
  const bf16x8 hi = *(const bf16x8*)(&Bs[n * LDSK + kb + 16]);
  return __builtin_shufflevector(lo, hi, 0, 1, 2, 3, 4, 5, 6, 7, 8, 9, 10, 11,
                                 12, 13, 14, 15);
}

DEV_INLINE f32x4 ld4(const float* p) { return *(const f32x4*)p; }

DEV_INLINE bf16x8 cvt8(f32x4 a, f32x4 b) {
  bf16x8 r;
  r[0] = (bf16)a[0]; r[1] = (bf16)a[1]; r[2] = (bf16)a[2]; r[3] = (bf16)a[3];
  r[4] = (bf16)b[0]; r[5] = (bf16)b[1]; r[6] = (bf16)b[2]; r[7] = (bf16)b[3];
  return r;
}

// load 16 contiguous-k A elements (addresses 16B aligned by construction)
template <int AMODE>
DEV_INLINE void load_a_vec16(const float* __restrict__ Ab, long long lda,
                             const float* __restrict__ aux,
                             const float* __restrict__ projv, int gm, int gk,
                             f32x4 v[4]) {
  const f32x4 zero = {};
  v[0] = zero; v[1] = zero; v[2] = zero; v[3] = zero;
  if (AMODE == A_SHIFT) {  // z_ctx: one-token shift, zeros at t==0
    if ((gm % kT) == 0) return;
    const float* p = Ab + (size_t)(gm - 1) * lda + gk;
    v[0] = ld4(p); v[1] = ld4(p + 4); v[2] = ld4(p + 8); v[3] = ld4(p + 12);
    return;
  }
  const float* p = Ab + (size_t)gm * lda + gk;
  v[0] = ld4(p); v[1] = ld4(p + 4); v[2] = ld4(p + 8); v[3] = ld4(p + 12);
  if (AMODE == A_SUBB) {           // x_input - b
    v[0] -= ld4(aux + gk);     v[1] -= ld4(aux + gk + 4);
    v[2] -= ld4(aux + gk + 8); v[3] -= ld4(aux + gk + 12);
  } else if (AMODE == A_COMBINE) { // z_novel + proj[m] * z_pred_
    const float pv = projv[gm];
    const float* q2 = aux + (size_t)gm * lda + gk;
    v[0] += pv * ld4(q2);     v[1] += pv * ld4(q2 + 4);
    v[2] += pv * ld4(q2 + 8); v[3] += pv * ld4(q2 + 12);
  }
}

// ---------------- generic batched WMMA GEMM ----------------
// C[m,n] = epilogue( alpha * sum_k A[m,k] * B[n,k] )
// A: k-contiguous rows (stride lda).
// BKC=true : B row-major (N,K), element Bb[n*bns + k]   (bks==1)
// BKC=false: B k-major   (K,N), element Bb[k*bks + n]   (bns==1)
template <int AMODE, bool BKC, bool BIAS, bool RELU>
__global__ __launch_bounds__(256) void tsae_wmma_gemm(
    const float* __restrict__ A, long long lda, long long a_ob, long long a_oh,
    const float* __restrict__ Bm, long long bks, long long bns, long long b_ob,
    long long b_oh, float* __restrict__ C, long long ldc, long long c_ob,
    long long c_oh, const float* __restrict__ biasv,
    const float* __restrict__ aux, const float* __restrict__ projv, int M,
    int N, int K, float alpha, int nh) {
  __shared__ __align__(16) bf16 As[BM * LDSK];
  __shared__ __align__(16) bf16 Bs[BN * LDSK];

  const int bb = blockIdx.z / nh;
  const int hh = blockIdx.z % nh;
  const float* Ab = A + (long long)bb * a_ob + (long long)hh * a_oh;
  const float* Bb = Bm + (long long)bb * b_ob + (long long)hh * b_oh;
  float* Cb = C + (long long)bb * c_ob + (long long)hh * c_oh;

  const int blockM = blockIdx.y * BM;
  const int blockN = blockIdx.x * BN;
  const int tid  = threadIdx.x;
  const int lane = tid & 31;
  const int wid  = tid >> 5;
  const int wm   = (wid >> 1) * 32;  // 4 waves along M
  const int wn   = (wid & 1) * 64;   // 2 waves along N

  // staging maps: 16 elements (one 64B global segment) per thread
  const int srow = tid >> 1;         // 0..127 (A row m / BKC n)
  const int scol = (tid & 1) * 16;   // 0 or 16 (k segment)
  // KN staging: 4x4 micro-transpose
  const int kq = (tid >> 5) * 4;     // 0,4,...,28
  const int nq = (tid & 31) * 4;     // 0..124

  v8f acc[2][4] = {};

  for (int kk = 0; kk < K; kk += TK) {
    // ---- stage A tile (BM x TK) [m][k], two aligned 16B LDS stores ----
    {
      const int gm = blockM + srow;
      const int gk = kk + scol;      // 16-aligned; all K are multiples of 16
      f32x4 v[4];
      if (gk < K) {
        load_a_vec16<AMODE>(Ab, lda, aux, projv, gm, gk, v);
      } else {
        const f32x4 z = {};
        v[0] = z; v[1] = z; v[2] = z; v[3] = z;
      }
      *(bf16x8*)(&As[srow * LDSK + scol])     = cvt8(v[0], v[1]);
      *(bf16x8*)(&As[srow * LDSK + scol + 8]) = cvt8(v[2], v[3]);
    }
    // ---- stage B tile as [n][k] ----
    if (BKC) {
      // weights (N,K) row-major: k-contiguous loads, k-contiguous stores
      const int gn = blockN + srow;
      const int gk = kk + scol;
      f32x4 w0 = {}, w1 = {}, w2 = {}, w3 = {};
      if (gk < K) {
        const float* p = Bb + (size_t)gn * bns + gk;
        w0 = ld4(p); w1 = ld4(p + 4); w2 = ld4(p + 8); w3 = ld4(p + 12);
        if (kk + TK < K) __builtin_prefetch(p + TK, 0, 0);
      }
      *(bf16x8*)(&Bs[srow * LDSK + scol])     = cvt8(w0, w1);
      *(bf16x8*)(&Bs[srow * LDSK + scol + 8]) = cvt8(w2, w3);
    } else {
      // (K,N) k-major: coalesced n-contiguous loads, 4x4 register transpose
      const int gk = kk + kq;
      f32x4 r0 = {}, r1 = {}, r2 = {}, r3 = {};
      if (gk < K) {
        const float* p = Bb + (size_t)gk * bks + blockN + nq;
        r0 = ld4(p);
        r1 = ld4(p + bks);
        r2 = ld4(p + 2 * (size_t)bks);
        r3 = ld4(p + 3 * (size_t)bks);
        if (kk + TK < K) __builtin_prefetch(p + (size_t)TK * bks, 0, 0);
      }
#pragma unroll
      for (int dn = 0; dn < 4; ++dn) {
        bf16x4 c;
        c[0] = (bf16)r0[dn]; c[1] = (bf16)r1[dn];
        c[2] = (bf16)r2[dn]; c[3] = (bf16)r3[dn];
        *(bf16x4*)(&Bs[(nq + dn) * LDSK + kq]) = c;
      }
    }
    __syncthreads();

    const v16bf af0 = load_a_frag(As, wm, lane);
    const v16bf af1 = load_a_frag(As, wm + 16, lane);
    v16bf bfr[4];
#pragma unroll
    for (int nt = 0; nt < 4; ++nt) bfr[nt] = load_b_frag(Bs, wn + nt * 16, lane);
#pragma unroll
    for (int nt = 0; nt < 4; ++nt) {
      acc[0][nt] = wmma_bf16(af0, bfr[nt], acc[0][nt]);
      acc[1][nt] = wmma_bf16(af1, bfr[nt], acc[1][nt]);
    }
    __syncthreads();
  }

  // ---- epilogue: 16x16 f32 C layout (VGPR v: lanes0-15 M=v, lanes16-31 M=v+8)
  const int lh  = lane >> 4;
  const int l16 = lane & 15;
#pragma unroll
  for (int mt = 0; mt < 2; ++mt)
#pragma unroll
    for (int nt = 0; nt < 4; ++nt)
#pragma unroll
      for (int v = 0; v < 8; ++v) {
        const int row = blockM + wm + mt * 16 + v + 8 * lh;
        const int col = blockN + wn + nt * 16 + l16;
        float val = acc[mt][nt][v] * alpha;
        if (BIAS) val += biasv[col];
        if (RELU) val = fmaxf(val, 0.0f);
        Cb[(size_t)row * ldc + col] = val;
      }
}

// ---------------- causal softmax: one wave32 per row of 512 ----------------
__global__ __launch_bounds__(256) void tsae_softmax_causal(float* __restrict__ sc) {
  const int wid  = threadIdx.x >> 5;
  const int lane = threadIdx.x & 31;
  const long long row = (long long)blockIdx.x * 8 + wid;  // over B*H*T rows
  const int i = (int)(row % kT);
  float* r = sc + row * kT;

  float vals[kT / 32];
  float mx = -3.4e38f;
#pragma unroll
  for (int jt = 0; jt < kT / 32; ++jt) {
    const int j = jt * 32 + lane;
    const float s = (j <= i) ? r[j] : -3.4e38f;
    vals[jt] = s;
    mx = fmaxf(mx, s);
  }
  for (int off = 16; off > 0; off >>= 1) mx = fmaxf(mx, __shfl_xor(mx, off, 32));

  float sum = 0.0f;
#pragma unroll
  for (int jt = 0; jt < kT / 32; ++jt) {
    const int j = jt * 32 + lane;
    const float e = (j <= i) ? __expf(vals[jt] - mx) : 0.0f;
    vals[jt] = e;
    sum += e;
  }
  for (int off = 16; off > 0; off >>= 1) sum += __shfl_xor(sum, off, 32);

  const float inv = 1.0f / sum;
#pragma unroll
  for (int jt = 0; jt < kT / 32; ++jt) r[jt * 32 + lane] = vals[jt] * inv;
}

// ------- proj = <Dz,x>/(<Dz,Dz>+eps); xwork = x - proj*Dz (x = x_in - b) ----
__global__ __launch_bounds__(256) void tsae_proj_update(
    const float* __restrict__ xin, const float* __restrict__ bvec,
    const float* __restrict__ Dz, float* __restrict__ proj,
    float* __restrict__ xwork) {
  const int wid  = threadIdx.x >> 5;
  const int lane = threadIdx.x & 31;
  const int row = blockIdx.x * 8 + wid;
  const float* xr  = xin + (size_t)row * kDIMIN;
  const float* dzr = Dz + (size_t)row * kDIMIN;

  float s1 = 0.0f, s2 = 0.0f;
  for (int d = lane; d < kDIMIN; d += 32) {
    const float xv = xr[d] - bvec[d];
    const float dz = dzr[d];
    s1 += dz * xv;
    s2 += dz * dz;
  }
  for (int off = 16; off > 0; off >>= 1) {
    s1 += __shfl_xor(s1, off, 32);
    s2 += __shfl_xor(s2, off, 32);
  }
  const float p = s1 / (s2 + kEPS);
  if (lane == 0) proj[row] = p;

  float* xw = xwork + (size_t)row * kDIMIN;
  for (int d = lane; d < kDIMIN; d += 32) {
    const float xv = xr[d] - bvec[d];
    xw[d] = xv - p * dzr[d];
  }
}

// --------- top-k(32) mask per token over WIDTH=8192, in place --------------
__global__ __launch_bounds__(256) void tsae_topk_mask(float* __restrict__ z,
                                                      const int* __restrict__ kptr) {
  const int row = blockIdx.x;
  float* zr = z + (size_t)row * kWIDTH;
  const int tid = threadIdx.x;

  float v[32], keep[32];
#pragma unroll
  for (int j = 0; j < 32; ++j) {
    v[j] = zr[tid + 256 * j];
    keep[j] = 0.0f;
  }

  __shared__ float wval[8];
  __shared__ int wtd[8];
  __shared__ int wjj[8];
  __shared__ int btid;
  __shared__ int bj;

  const int kv = kptr[0];
  for (int it = 0; it < kv; ++it) {
    float bv = -1.0f;  // relu outputs are >= 0
    int bl = 0;
#pragma unroll
    for (int j = 0; j < 32; ++j)
      if (v[j] > bv) { bv = v[j]; bl = j; }
    int bt = tid;
    for (int off = 16; off > 0; off >>= 1) {
      const float ov = __shfl_xor(bv, off, 32);
      const int ot = __shfl_xor(bt, off, 32);
      const int ol = __shfl_xor(bl, off, 32);
      if (ov > bv || (ov == bv && ot < bt)) { bv = ov; bt = ot; bl = ol; }
    }
    const int wi = tid >> 5;
    if ((tid & 31) == 0) { wval[wi] = bv; wtd[wi] = bt; wjj[wi] = bl; }
    __syncthreads();
    if (tid == 0) {
      float xb = wval[0];
      int xt = wtd[0], xj = wjj[0];
      for (int w = 1; w < 8; ++w)
        if (wval[w] > xb || (wval[w] == xb && wtd[w] < xt)) {
          xb = wval[w]; xt = wtd[w]; xj = wjj[w];
        }
      btid = xt;
      bj = xj;
    }
    __syncthreads();
    if (tid == btid) { keep[bj] = v[bj]; v[bj] = -2.0f; }
    __syncthreads();
  }
#pragma unroll
  for (int j = 0; j < 32; ++j) zr[tid + 256 * j] = keep[j];
}

// ---------------- host-side orchestration ----------------
extern "C" void kernel_launch(void* const* d_in, const int* in_sizes, int n_in,
                              void* d_out, int out_size, void* d_ws,
                              size_t ws_size, hipStream_t stream) {
  (void)in_sizes; (void)n_in; (void)out_size; (void)ws_size;
  const float* x_in = (const float*)d_in[0];
  const float* D    = (const float*)d_in[1];
  const float* bvec = (const float*)d_in[2];
  const float* Wk   = (const float*)d_in[3];
  const float* bk   = (const float*)d_in[4];
  const float* Wq   = (const float*)d_in[5];
  const float* bq   = (const float*)d_in[6];
  const float* Wv   = (const float*)d_in[7];
  const float* bv   = (const float*)d_in[8];
  const float* Wo   = (const float*)d_in[9];
  const float* bo   = (const float*)d_in[10];
  const int*   kvp  = (const int*)d_in[11];
  float* out = (float*)d_out;

  float* w = (float*)d_ws;
  float* zin  = w;                                      // (4096,8192); reused as attn-o
  float* vbuf = zin + (size_t)kNTOK * kWIDTH;           // v; reused as z_novel
  float* zp   = vbuf + (size_t)kNTOK * kWIDTH;          // z_pred_
  float* sc   = zp + (size_t)kNTOK * kWIDTH;            // (B,H,T,T) scores/attn
  float* qb   = sc + (size_t)kB * kNHEADS * kT * kT;    // (4096,128)
  float* kb_  = qb + (size_t)kNTOK * kNEMBDS;           // (4096,128)
  float* dz   = kb_ + (size_t)kNTOK * kNEMBDS;          // (4096,768)
  float* pr   = dz + (size_t)kNTOK * kDIMIN;            // (4096)
  float* xw   = pr + kNTOK;                             // (4096,768)

  const dim3 blk(256);
  const long long ZERO = 0;

  // 1) z_in = relu(lam * (x - b) @ D^T)
  tsae_wmma_gemm<A_SUBB, true, false, true>
      <<<dim3(kWIDTH / BN, kNTOK / BM, 1), blk, 0, stream>>>(
          x_in, kDIMIN, ZERO, ZERO, D, 1, kDIMIN, ZERO, ZERO, zin, kWIDTH,
          ZERO, ZERO, nullptr, bvec, nullptr, kNTOK, kWIDTH, kDIMIN, kLAM, 1);
  // 2) q = z_in @ Wq^T + bq
  tsae_wmma_gemm<A_PLAIN, true, true, false>
      <<<dim3(kNEMBDS / BN, kNTOK / BM, 1), blk, 0, stream>>>(
          zin, kWIDTH, ZERO, ZERO, Wq, 1, kWIDTH, ZERO, ZERO, qb, kNEMBDS,
          ZERO, ZERO, bq, nullptr, nullptr, kNTOK, kNEMBDS, kWIDTH, 1.0f, 1);
  // 3) k = z_ctx @ Wk^T + bk  (shift fused into A-loader)
  tsae_wmma_gemm<A_SHIFT, true, true, false>
      <<<dim3(kNEMBDS / BN, kNTOK / BM, 1), blk, 0, stream>>>(
          zin, kWIDTH, ZERO, ZERO, Wk, 1, kWIDTH, ZERO, ZERO, kb_, kNEMBDS,
          ZERO, ZERO, bk, nullptr, nullptr, kNTOK, kNEMBDS, kWIDTH, 1.0f, 1);
  // 4) v = z_ctx @ Wv^T + bv
  tsae_wmma_gemm<A_SHIFT, true, true, false>
      <<<dim3(kWIDTH / BN, kNTOK / BM, 1), blk, 0, stream>>>(
          zin, kWIDTH, ZERO, ZERO, Wv, 1, kWIDTH, ZERO, ZERO, vbuf, kWIDTH,
          ZERO, ZERO, bv, nullptr, nullptr, kNTOK, kWIDTH, kWIDTH, 1.0f, 1);
  // 5) scores = 0.25 * q @ k^T, batched over (b,h)
  tsae_wmma_gemm<A_PLAIN, true, false, false>
      <<<dim3(kT / BN, kT / BM, kB * kNHEADS), blk, 0, stream>>>(
          qb, kNEMBDS, (long long)kT * kNEMBDS, kDHQK, kb_, 1, kNEMBDS,
          (long long)kT * kNEMBDS, kDHQK, sc, kT,
          (long long)kNHEADS * kT * kT, (long long)kT * kT, nullptr, nullptr,
          nullptr, kT, kT, kDHQK, 0.25f, kNHEADS);
  // 6) causal softmax over rows of sc
  tsae_softmax_causal<<<dim3(kB * kNHEADS * kT / 8), blk, 0, stream>>>(sc);
  // 7) o = a @ v_head  -> stored into zin (no longer needed)
  tsae_wmma_gemm<A_PLAIN, false, false, false>
      <<<dim3(kDHV / BN, kT / BM, kB * kNHEADS), blk, 0, stream>>>(
          sc, kT, (long long)kNHEADS * kT * kT, (long long)kT * kT, vbuf,
          kWIDTH, 1, (long long)kT * kWIDTH, kDHV, zin, kWIDTH,
          (long long)kT * kWIDTH, kDHV, nullptr, nullptr, nullptr, kT, kDHV,
          kT, 1.0f, kNHEADS);
  // 8) z_pred_ = relu(o @ Wo^T + bo)
  tsae_wmma_gemm<A_PLAIN, true, true, true>
      <<<dim3(kWIDTH / BN, kNTOK / BM, 1), blk, 0, stream>>>(
          zin, kWIDTH, ZERO, ZERO, Wo, 1, kWIDTH, ZERO, ZERO, zp, kWIDTH,
          ZERO, ZERO, bo, nullptr, nullptr, kNTOK, kWIDTH, kWIDTH, 1.0f, 1);
  // 9) Dz = z_pred_ @ D  (D used K-major)
  tsae_wmma_gemm<A_PLAIN, false, false, false>
      <<<dim3(kDIMIN / BN, kNTOK / BM, 1), blk, 0, stream>>>(
          zp, kWIDTH, ZERO, ZERO, D, kDIMIN, 1, ZERO, ZERO, dz, kDIMIN, ZERO,
          ZERO, nullptr, nullptr, nullptr, kNTOK, kDIMIN, kWIDTH, 1.0f, 1);
  // 10) proj + residual update
  tsae_proj_update<<<dim3(kNTOK / 8), blk, 0, stream>>>(x_in, bvec, dz, pr, xw);
  // 11) z_novel = relu(lam * xw @ D^T) -> vbuf
  tsae_wmma_gemm<A_PLAIN, true, false, true>
      <<<dim3(kWIDTH / BN, kNTOK / BM, 1), blk, 0, stream>>>(
          xw, kDIMIN, ZERO, ZERO, D, 1, kDIMIN, ZERO, ZERO, vbuf, kWIDTH,
          ZERO, ZERO, nullptr, nullptr, nullptr, kNTOK, kWIDTH, kDIMIN, kLAM, 1);
  // 12) top-k(32) sparsification in place
  tsae_topk_mask<<<dim3(kNTOK), blk, 0, stream>>>(vbuf, kvp);
  // 13) x_recons = (z_novel + proj*z_pred_) @ D + b  (A-combine fused)
  tsae_wmma_gemm<A_COMBINE, false, true, false>
      <<<dim3(kDIMIN / BN, kNTOK / BM, 1), blk, 0, stream>>>(
          vbuf, kWIDTH, ZERO, ZERO, D, kDIMIN, 1, ZERO, ZERO, out, kDIMIN,
          ZERO, ZERO, bvec, zp, pr, kNTOK, kDIMIN, kWIDTH, 1.0f, 1);
}
